// XTransEncoderActrag_79963701117004
// MI455X (gfx1250) — matrix operand order repaired and checked by reading, not compiled, not run on hardware
//
#include <hip/hip_runtime.h>
#include <math.h>

// EPT layer dims
#define B_ 4
#define N_ 1024
#define P_ 128
#define D_ 512
#define NH_ 8
#define DH_ 64

typedef __attribute__((ext_vector_type(16))) __bf16 v16bf;
typedef __attribute__((ext_vector_type(8)))  __bf16 v8bf;
typedef __attribute__((ext_vector_type(8)))  float  v8f;
typedef __attribute__((ext_vector_type(4)))  unsigned v4u;
typedef __attribute__((ext_vector_type(8)))  int      v8i;
typedef __attribute__((ext_vector_type(4)))  int      v4i;

static __device__ __forceinline__ __bf16 f2bf(float f) {
  unsigned u = __builtin_bit_cast(unsigned, f);
  unsigned r = u + 0x7FFFu + ((u >> 16) & 1u);   // round-to-nearest-even
  unsigned short h = (unsigned short)(r >> 16);
  return __builtin_bit_cast(__bf16, h);
}

// concat two 8-lane bf16 vectors (two 16B loads) into a WMMA operand
static __device__ __forceinline__ v16bf ld16(const __bf16* p0, const __bf16* p1) {
  v8bf a = *(const v8bf*)p0;
  v8bf b = *(const v8bf*)p1;
  return __builtin_shufflevector(a, b, 0, 1, 2, 3, 4, 5, 6, 7, 8, 9, 10, 11, 12, 13, 14, 15);
}

// ---------------------------------------------------------------------------
// TDM: DMA a 64-row x 32-col bf16 tile of W (row stride K elements) into LDS
// at byte offset ldsByte. 2-D descriptor per cdna5 ISA 08_async_tensor §8.
// ---------------------------------------------------------------------------
static __device__ __forceinline__ void tdm_load_w_tile(const __bf16* gptr,
                                                       unsigned ldsByte, int K) {
  unsigned long long ga = (unsigned long long)(size_t)gptr;
  v4u g0;
  g0[0] = 1u;                                   // count=1, user mode, no gather
  g0[1] = ldsByte;                              // lds_addr [63:32]
  g0[2] = (unsigned)(ga & 0xFFFFFFFFu);         // global_addr lo
  g0[3] = (unsigned)((ga >> 32) & 0x1FFFFFFu)   // global_addr hi (57-bit)
          | (2u << 30);                         // type=2 ("image")
  const unsigned td0 = (unsigned)K;             // tensor_dim0 (elements)
  v8i g1;
  g1[0] = (int)0x00010000u;                     // data_size=1 (2 bytes)
  g1[1] = (int)((td0 & 0xFFFFu) << 16);         // tensor_dim0[15:0] @ bits 63:48
  g1[2] = (int)((td0 >> 16) | (64u << 16));     // tensor_dim0[31:16] | tensor_dim1=64
  g1[3] = (int)(32u << 16);                     // tensor_dim1 hi | tile_dim0=32
  g1[4] = 64;                                   // tile_dim1=64, tile_dim2=0
  g1[5] = (int)td0;                             // tensor_dim0_stride lo32 = K
  g1[6] = 0;                                    // stride hi | dim1_stride lo
  g1[7] = 0;
  const v4i z4 = {0, 0, 0, 0};
  const v8i z8 = {0, 0, 0, 0, 0, 0, 0, 0};
  __builtin_amdgcn_tensor_load_to_lds(g0, g1, z4, z4, z8, 0);
}

// ---------------------------------------------------------------------------
// Generic GEMM: C = act(A[M,K]bf16 @ W[N,K]bf16^T + bias) + resid
// W tiles staged into LDS by the Tensor Data Mover, double buffered.
// dual outputs: Cf (f32, optional) and Cbf (bf16, optional)
// grid = (N/64, M/64), block = 128 (4 waves), dynamic LDS = 8192 B.
// ACT: 0 none, 1 SiLU
// ---------------------------------------------------------------------------
template <int ACT>
__global__ void gemm_wmma(const __bf16* __restrict__ A, const __bf16* __restrict__ W,
                          const float* __restrict__ bias, const float* __restrict__ resid,
                          float* __restrict__ Cf, __bf16* __restrict__ Cbf,
                          int M, int N, int K) {
  extern __shared__ __bf16 wlds[];  // 2 x (64 rows x 32 k) bf16 = 8192 B
  const int lane  = threadIdx.x & 31;
  const int wave  = threadIdx.x >> 5;
  const int l15   = lane & 15;
  const int khalf = lane >> 4;
  const int m     = blockIdx.y * 64 + wave * 16 + l15;  // A fragment row
  const int ncb   = blockIdx.x * 64;

  v8f acc[4];
#pragma unroll
  for (int t = 0; t < 4; ++t)
#pragma unroll
    for (int r = 0; r < 8; ++r) acc[t][r] = 0.f;

  const __bf16* arow = A + (size_t)m * K + khalf * 8;
  const __bf16* wpan = W + (size_t)ncb * K;  // 64-row W panel for this block
  const int nk = K >> 5;

  if (wave == 0) tdm_load_w_tile(wpan, 0u, K);  // prefetch tile 0 -> buf 0

  for (int i = 0; i < nk; ++i) {
    const int cur = i & 1;
    if (wave == 0) {
      if (i + 1 < nk) {
        tdm_load_w_tile(wpan + (i + 1) * 32, (unsigned)(((i + 1) & 1) * 4096), K);
        __builtin_amdgcn_s_wait_tensorcnt(1);  // tile i complete (in-order)
      } else {
        __builtin_amdgcn_s_wait_tensorcnt(0);
      }
    }
    __syncthreads();  // publish buf[cur] to all waves

    // A frag: lane = row m, K = i*32 + khalf*8 + {0..7, 16..23} (global)
    v16bf a = ld16(arow + i * 32, arow + i * 32 + 16);
#pragma unroll
    for (int t = 0; t < 4; ++t) {
      // B frag from LDS: row (t*16+l15) of tile, K = khalf*8 + {0..7, 16..23}
      const __bf16* wl = wlds + cur * 2048 + (t * 16 + l15) * 32 + khalf * 8;
      v16bf bf = ld16(wl, wl + 16);
      acc[t] = __builtin_amdgcn_wmma_f32_16x16x32_bf16(false, a, false, bf,
                                                       (short)0, acc[t], false, false);
    }
    __syncthreads();  // all reads of buf[cur] done before TDM overwrites it
  }

  const int mr0 = blockIdx.y * 64 + wave * 16 + 8 * khalf;
#pragma unroll
  for (int t = 0; t < 4; ++t) {
    const int col = ncb + t * 16 + l15;
    const float bv = bias ? bias[col] : 0.f;
#pragma unroll
    for (int r = 0; r < 8; ++r) {
      const int row = mr0 + r;
      float v = acc[t][r] + bv;
      if (ACT == 1) v = v / (1.f + __expf(-v));  // SiLU
      if (resid) v += resid[(size_t)row * N + col];
      if (Cf)  Cf[(size_t)row * N + col] = v;
      if (Cbf) Cbf[(size_t)row * N + col] = f2bf(v);
    }
  }
}

// ---------------------------------------------------------------------------
// Flash attention: one wave per 16 query rows, head h, batch b.
// Q/K bf16 (token-major, row stride qs/ks, head offset h*DQ).
// V bf16 TRANSPOSED: (B, NH, DVAL, n_kv).  O f32 (token-major, stride os).
// grid = (n_q/16, NH, B), block = 32.
// ---------------------------------------------------------------------------
template <int DQ, int DVAL, bool HASB>
__global__ void flash_attn(const __bf16* __restrict__ Q, int qs,
                           const __bf16* __restrict__ Kt, int ks,
                           const __bf16* __restrict__ Vt,
                           float* __restrict__ O, int os,
                           const float* __restrict__ Db, const float* __restrict__ rbf,
                           const int* __restrict__ msk,
                           int n_q, int n_kv, float scale) {
  constexpr int NKF = DQ / 32;
  constexpr int NT  = DVAL / 16;
  const int lane  = threadIdx.x;
  const int l15   = lane & 15;
  const int khalf = lane >> 4;
  const int h = blockIdx.y, b = blockIdx.z;
  const int q0 = blockIdx.x * 16;

  v16bf qf[NKF];
  {
    const __bf16* qrow = Q + (size_t)(b * n_q + q0 + l15) * qs + h * DQ + khalf * 8;
#pragma unroll
    for (int f = 0; f < NKF; ++f)
      qf[f] = ld16(qrow + f * 32, qrow + f * 32 + 16);
  }

  v8f o[NT];
#pragma unroll
  for (int t = 0; t < NT; ++t)
#pragma unroll
    for (int r = 0; r < 8; ++r) o[t][r] = 0.f;

  float rm[8], rl[8];
#pragma unroll
  for (int r = 0; r < 8; ++r) { rm[r] = -1e30f; rl[r] = 0.f; }

  __shared__ __bf16 pls[16 * 32];  // P tile bounce (C-layout -> A-layout)

  const __bf16* vbase = Vt + ((size_t)(b * NH_ + h) * DVAL) * n_kv;

  for (int k0 = 0; k0 < n_kv; k0 += 32) {
    float sv[2][8];
#pragma unroll
    for (int hf = 0; hf < 2; ++hf) {
      const int kcol = k0 + hf * 16 + l15;
      const __bf16* krow = Kt + (size_t)(b * n_kv + kcol) * ks + h * DQ + khalf * 8;
      v8f s;
#pragma unroll
      for (int r = 0; r < 8; ++r) s[r] = 0.f;
#pragma unroll
      for (int f = 0; f < NKF; ++f) {
        v16bf kb = ld16(krow + f * 32, krow + f * 32 + 16);
        s = __builtin_amdgcn_wmma_f32_16x16x32_bf16(false, qf[f], false, kb,
                                                    (short)0, s, false, false);
      }
#pragma unroll
      for (int r = 0; r < 8; ++r) {
        float x = s[r] * scale;
        if (HASB) {
          const int qi = q0 + r + 8 * khalf;
          const int ki = k0 + hf * 16 + l15;
          const size_t bi = ((size_t)b * n_q + qi) * n_kv + ki;
          x += Db[bi] + rbf[bi];
          if (msk[b * n_kv + ki] == 0) x = -1e30f;
        }
        sv[hf][r] = x;
      }
    }
    float cm[8];
#pragma unroll
    for (int r = 0; r < 8; ++r) cm[r] = fmaxf(sv[0][r], sv[1][r]);
#pragma unroll
    for (int off = 1; off < 16; off <<= 1)
#pragma unroll
      for (int r = 0; r < 8; ++r) cm[r] = fmaxf(cm[r], __shfl_xor(cm[r], off, 32));

    float alpha[8], ps[8];
#pragma unroll
    for (int r = 0; r < 8; ++r) {
      const float mn = fmaxf(rm[r], cm[r]);
      alpha[r] = __expf(rm[r] - mn);
      rm[r] = mn;
      ps[r] = 0.f;
    }
#pragma unroll
    for (int hf = 0; hf < 2; ++hf)
#pragma unroll
      for (int r = 0; r < 8; ++r) {
        const float p = __expf(sv[hf][r] - rm[r]);
        sv[hf][r] = p;
        ps[r] += p;
      }
#pragma unroll
    for (int off = 1; off < 16; off <<= 1)
#pragma unroll
      for (int r = 0; r < 8; ++r) ps[r] += __shfl_xor(ps[r], off, 32);
#pragma unroll
    for (int r = 0; r < 8; ++r) rl[r] = rl[r] * alpha[r] + ps[r];
#pragma unroll
    for (int t = 0; t < NT; ++t)
#pragma unroll
      for (int r = 0; r < 8; ++r) o[t][r] *= alpha[r];

    __syncthreads();
#pragma unroll
    for (int hf = 0; hf < 2; ++hf)
#pragma unroll
      for (int r = 0; r < 8; ++r)
        pls[(r + 8 * khalf) * 32 + hf * 16 + l15] = f2bf(sv[hf][r]);
    __syncthreads();
    v16bf pf = ld16(&pls[l15 * 32 + khalf * 8], &pls[l15 * 32 + khalf * 8 + 16]);

#pragma unroll
    for (int t = 0; t < NT; ++t) {
      const __bf16* vrow = vbase + (size_t)(t * 16 + l15) * n_kv + k0 + khalf * 8;
      v16bf vb = ld16(vrow, vrow + 16);
      o[t] = __builtin_amdgcn_wmma_f32_16x16x32_bf16(false, pf, false, vb,
                                                     (short)0, o[t], false, false);
    }
  }

  const int qr0 = q0 + 8 * khalf;
#pragma unroll
  for (int t = 0; t < NT; ++t) {
    const int col = h * DVAL + t * 16 + l15;
#pragma unroll
    for (int r = 0; r < 8; ++r) {
      const float den = rl[r];
      const float val = (den > 0.f) ? o[t][r] / den : 0.f;
      O[(size_t)(b * n_q + qr0 + r) * os + col] = val;
    }
  }
}

// ---------------------------------------------------------------------------
// LayerNorm over D=512 (f32 in, bf16 out), one 256-thr block per row.
// ---------------------------------------------------------------------------
__global__ void ln_kernel(const float* __restrict__ x, const float* __restrict__ g,
                          const float* __restrict__ bta, __bf16* __restrict__ out,
                          int outStride) {
  const int row = blockIdx.x, tid = threadIdx.x;
  const float* xr = x + (size_t)row * 512;
  const float v0 = xr[tid], v1 = xr[tid + 256];
  __shared__ float s1[256], s2[256];
  s1[tid] = v0 + v1;
  s2[tid] = v0 * v0 + v1 * v1;
  __syncthreads();
  for (int off = 128; off > 0; off >>= 1) {
    if (tid < off) { s1[tid] += s1[tid + off]; s2[tid] += s2[tid + off]; }
    __syncthreads();
  }
  const float mean = s1[0] * (1.f / 512.f);
  const float var  = s2[0] * (1.f / 512.f) - mean * mean;
  const float rstd = rsqrtf(var + 1e-5f);
  __bf16* orow = out + (size_t)row * outStride;
  orow[tid]       = f2bf((v0 - mean) * rstd * g[tid] + bta[tid]);
  orow[tid + 256] = f2bf((v1 - mean) * rstd * g[tid + 256] + bta[tid + 256]);
}

__global__ void cvt_f32_bf16(const float* __restrict__ in, __bf16* __restrict__ out,
                             int total) {
  const int i = blockIdx.x * blockDim.x + threadIdx.x;
  if (i < total) out[i] = f2bf(in[i]);
}

// V_attn transposed bf16: Va_t[((b*NH+h)*256 + d)*N + n]
__global__ void assemble_vattn_t(const float* __restrict__ Hv, const float* __restrict__ Vv,
                                 __bf16* __restrict__ Va, int total) {
  const int i = blockIdx.x * blockDim.x + threadIdx.x;
  if (i >= total) return;
  const int n = i & 1023, d = (i >> 10) & 255, hh = (i >> 18) & 7, b = i >> 21;
  const int tok = b * 1024 + n;
  float v;
  if (d < 64) v = Hv[(size_t)tok * 512 + hh * 64 + d];
  else {
    const int c = (d - 64) >> 6, dd = (d - 64) & 63;
    v = Vv[((size_t)tok * 3 + c) * 512 + hh * 64 + dd];
  }
  Va[i] = f2bf(v);
}

__global__ void disassemble_res(const float* __restrict__ res, __bf16* __restrict__ Hres,
                                __bf16* __restrict__ Vres, int total) {
  const int i = blockIdx.x * blockDim.x + threadIdx.x;
  if (i >= total) return;
  const int d = i & 255, hh = (i >> 8) & 7, tok = i >> 11;
  const __bf16 v = f2bf(res[i]);
  if (d < 64) Hres[(size_t)tok * 512 + hh * 64 + d] = v;
  else {
    const int c = (d - 64) >> 6, dd = (d - 64) & 63;
    Vres[((size_t)tok * 3 + c) * 512 + hh * 64 + dd] = v;
  }
}

// vp (B*P,512) bf16 -> vpt[((b*NH+h)*64 + d)*128 + p]
__global__ void transpose_vp(const __bf16* __restrict__ in, __bf16* __restrict__ out,
                             int total) {
  const int i = blockIdx.x * blockDim.x + threadIdx.x;
  if (i >= total) return;
  const int p = i & 127, d = (i >> 7) & 63, hh = (i >> 13) & 7, b = i >> 16;
  out[i] = in[(size_t)(b * 128 + p) * 512 + hh * 64 + d];
}

// scaler_bf[tok,512+d] = || Vp[tok, 0:3, d] ||
__global__ void vnorm_kernel(const float* __restrict__ Vp, __bf16* __restrict__ scaler,
                             int total) {
  const int i = blockIdx.x * blockDim.x + threadIdx.x;
  if (i >= total) return;
  const int d = i & 511, tok = i >> 9;
  float s = 0.f;
#pragma unroll
  for (int c = 0; c < 3; ++c) {
    const float x = Vp[((size_t)tok * 3 + c) * 1024 + d];
    s += x * x;
  }
  scaler[(size_t)tok * 1024 + 512 + d] = f2bf(sqrtf(s));
}

// H = so[:,:512] + H2 ; V = so[:,512:][...,None,:] * Vp[...,512:] + V0
__global__ void final_epilogue(const float* __restrict__ so, const float* __restrict__ H2,
                               const float* __restrict__ Vp, const float* __restrict__ V0,
                               float* __restrict__ outH, float* __restrict__ outV,
                               int total) {
  const int i = blockIdx.x * blockDim.x + threadIdx.x;
  if (i >= total) return;
  const int d = i & 511, tok = i >> 9;
  outH[i] = so[(size_t)tok * 1024 + d] + H2[i];
  const float vu = so[(size_t)tok * 1024 + 512 + d];
#pragma unroll
  for (int c = 0; c < 3; ++c) {
    const size_t vi = ((size_t)tok * 3 + c) * 512 + d;
    outV[vi] = vu * Vp[((size_t)tok * 3 + c) * 1024 + 512 + d] + V0[vi];
  }
}

// ---------------------------------------------------------------------------
extern "C" void kernel_launch(void* const* d_in, const int* in_sizes, int n_in,
                              void* d_out, int out_size, void* d_ws, size_t ws_size,
                              hipStream_t stream) {
  (void)in_sizes; (void)n_in; (void)out_size; (void)ws_size;
  const float* H_in  = (const float*)d_in[0];
  const float* V_in  = (const float*)d_in[1];
  const float* Db    = (const float*)d_in[2];
  const float* rbf   = (const float*)d_in[3];
  const float* prm   = (const float*)d_in[4];
  const float* Wq    = (const float*)d_in[5];
  const float* bq    = (const float*)d_in[6];
  const float* Wk    = (const float*)d_in[7];
  const float* bk    = (const float*)d_in[8];
  const float* Wv    = (const float*)d_in[9];
  const float* bv    = (const float*)d_in[10];
  const float* Wvv   = (const float*)d_in[11];
  const float* Wo    = (const float*)d_in[12];
  const float* bo    = (const float*)d_in[13];
  const float* Wvo   = (const float*)d_in[14];
  const float* ln1g  = (const float*)d_in[15];
  const float* ln1b  = (const float*)d_in[16];
  const float* cainw = (const float*)d_in[17];
  const float* cainb = (const float*)d_in[18];
  const float* caow  = (const float*)d_in[19];
  const float* caob  = (const float*)d_in[20];
  const float* casw  = (const float*)d_in[21];
  const float* casb  = (const float*)d_in[22];
  const float* ln2g  = (const float*)d_in[23];
  const float* ln2b  = (const float*)d_in[24];
  const float* flv   = (const float*)d_in[25];
  const float* fw1   = (const float*)d_in[26];
  const float* fb1   = (const float*)d_in[27];
  const float* fw2   = (const float*)d_in[28];
  const float* fb2   = (const float*)d_in[29];
  const float* ln3g  = (const float*)d_in[30];
  const float* ln3b  = (const float*)d_in[31];
  const int*   hmask = (const int*)d_in[32];

  // ---------------- f32 workspace (26M floats, stage-aliased) ----------------
  float* wsf = (float*)d_ws;
  const size_t MF = 1u << 20;
  float* Hv  = wsf;             // 2M  (stage A)
  float* Vv  = wsf + 2 * MF;    // 6M  (stage A)
  float* res = wsf + 8 * MF;    // 8M  (stage A)
  float* H1  = wsf + 16 * MF;   // 2M  (A->B)
  float* V1b = wsf + 18 * MF;   // 6M  (A->C, epilogue V0)
  float* H2  = wsf + 24 * MF;   // 2M  (B->C)
  float* co  = Hv;              // 2M  (stage B; Hv dead)
  float* Vp  = wsf + 2 * MF;    // 12M (stage C; Vv+res dead)
  float* so  = wsf + 14 * MF;   // 4M  (stage C; res tail + H1 dead)

  // ---------------- bf16 workspace (73.75M elems) ----------------
  __bf16* wsb = (__bf16*)(wsf + 26 * MF);
  const size_t Q4 = MF / 4;  // 262144
  __bf16* wq_b   = wsb;             // 4Q
  __bf16* wk_b   = wsb + 4  * Q4;   // 4Q
  __bf16* wv_b   = wsb + 8  * Q4;   // 1Q
  __bf16* wvv_b  = wsb + 9  * Q4;   // 1Q
  __bf16* wo_b   = wsb + 10 * Q4;   // 1Q
  __bf16* wvo_b  = wsb + 11 * Q4;   // 1Q
  __bf16* cain_b = wsb + 12 * Q4;   // 3Q
  __bf16* caow_b = wsb + 15 * Q4;   // 1Q
  __bf16* casw_b = wsb + 16 * Q4;   // 1Q
  __bf16* flv_b  = wsb + 17 * Q4;   // 2Q
  __bf16* fw1_b  = wsb + 19 * Q4;   // 8Q
  __bf16* fw2_b  = wsb + 27 * Q4;   // 8Q
  __bf16* Hn_b   = wsb + 35 * Q4;   // 8Q
  __bf16* Hq_b   = wsb + 43 * Q4;   // 32Q
  __bf16* Hk_b   = wsb + 75 * Q4;   // 32Q
  __bf16* Va_b   = wsb + 107 * Q4;  // 32Q  (transposed V_attn)
  __bf16* Hres_b = wsb + 139 * Q4;  // 8Q
  __bf16* Vres_b = wsb + 147 * Q4;  // 24Q
  __bf16* Vin_b  = wsb + 171 * Q4;  // 24Q
  __bf16* prm_b  = wsb + 195 * Q4;  // 1Q
  __bf16* qb_b   = wsb + 196 * Q4;  // 8Q
  __bf16* kp_b   = wsb + 204 * Q4;  // 1Q
  __bf16* vp_b   = wsb + 205 * Q4;  // 1Q
  __bf16* vpt_b  = wsb + 206 * Q4;  // 1Q  (transposed)
  __bf16* co_b   = wsb + 207 * Q4;  // 8Q
  __bf16* co2_b  = wsb + 215 * Q4;  // 8Q
  __bf16* scal_b = wsb + 223 * Q4;  // 16Q
  __bf16* h1_b   = wsb + 239 * Q4;  // 32Q
  __bf16* V1b_b  = wsb + 271 * Q4;  // 24Q

  float* outH = (float*)d_out;
  float* outV = (float*)d_out + 2 * MF;

  auto CVT = [&](const float* s, __bf16* t, int n) {
    cvt_f32_bf16<<<(n + 255) / 256, 256, 0, stream>>>(s, t, n);
  };
  const size_t LDSB = 8192;  // GEMM dynamic LDS (double-buffered W tile)

  // one-time bf16 operand conversions (weights + raw inputs)
  CVT(Wq, wq_b, 1048576);   CVT(Wk, wk_b, 1048576);
  CVT(Wv, wv_b, 262144);    CVT(Wvv, wvv_b, 262144);
  CVT(Wo, wo_b, 262144);    CVT(Wvo, wvo_b, 262144);
  CVT(cainw, cain_b, 786432);
  CVT(caow, caow_b, 262144); CVT(casw, casw_b, 262144);
  CVT(flv, flv_b, 524288);
  CVT(fw1, fw1_b, 2097152);  CVT(fw2, fw2_b, 2097152);
  CVT(V_in, Vin_b, 6291456); CVT(prm, prm_b, 262144);

  // ---- Stage A: self attention over H/V ----
  ln_kernel<<<4096, 256, 0, stream>>>(H_in, ln1g, ln1b, Hn_b, 512);
  gemm_wmma<0><<<dim3(32, 64), 128, LDSB, stream>>>(Hn_b, wq_b, bq, nullptr, nullptr, Hq_b, 4096, 2048, 512);
  gemm_wmma<0><<<dim3(32, 64), 128, LDSB, stream>>>(Hn_b, wk_b, bk, nullptr, nullptr, Hk_b, 4096, 2048, 512);
  gemm_wmma<0><<<dim3(8, 64), 128, LDSB, stream>>>(Hn_b, wv_b, bv, nullptr, Hv, nullptr, 4096, 512, 512);
  gemm_wmma<0><<<dim3(8, 192), 128, LDSB, stream>>>(Vin_b, wvv_b, nullptr, nullptr, Vv, nullptr, 12288, 512, 512);
  assemble_vattn_t<<<32768, 256, 0, stream>>>(Hv, Vv, Va_b, 8388608);
  flash_attn<256, 256, true><<<dim3(64, 8, 4), 32, 0, stream>>>(
      Hq_b, 2048, Hk_b, 2048, Va_b, res, 2048, Db, rbf, hmask, 1024, 1024, 0.0625f);
  disassemble_res<<<32768, 256, 0, stream>>>(res, Hres_b, Vres_b, 8388608);
  gemm_wmma<0><<<dim3(8, 64), 128, LDSB, stream>>>(Hres_b, wo_b, bo, H_in, H1, nullptr, 4096, 512, 512);
  gemm_wmma<0><<<dim3(8, 192), 128, LDSB, stream>>>(Vres_b, wvo_b, nullptr, V_in, V1b, V1b_b, 12288, 512, 512);

  // ---- Stage B: cross attention to prompt ----
  ln_kernel<<<4096, 256, 0, stream>>>(H1, ln2g, ln2b, Hn_b, 512);
  gemm_wmma<0><<<dim3(8, 64), 128, LDSB, stream>>>(Hn_b, cain_b, cainb, nullptr, nullptr, qb_b, 4096, 512, 512);
  gemm_wmma<0><<<dim3(8, 8), 128, LDSB, stream>>>(prm_b, cain_b + 512 * 512, cainb + 512, nullptr, nullptr, kp_b, 512, 512, 512);
  gemm_wmma<0><<<dim3(8, 8), 128, LDSB, stream>>>(prm_b, cain_b + 2 * 512 * 512, cainb + 1024, nullptr, nullptr, vp_b, 512, 512, 512);
  transpose_vp<<<1024, 256, 0, stream>>>(vp_b, vpt_b, 262144);
  flash_attn<64, 64, false><<<dim3(64, 8, 4), 32, 0, stream>>>(
      qb_b, 512, kp_b, 512, vpt_b, co, 512, nullptr, nullptr, nullptr, 1024, 128, 0.125f);
  CVT(co, co_b, 2097152);
  gemm_wmma<0><<<dim3(8, 64), 128, LDSB, stream>>>(co_b, caow_b, caob, nullptr, nullptr, co2_b, 4096, 512, 512);
  gemm_wmma<0><<<dim3(8, 64), 128, LDSB, stream>>>(co2_b, casw_b, casb, H1, H2, nullptr, 4096, 512, 512);

  // ---- Stage C: gated FFN ----
  ln_kernel<<<4096, 256, 0, stream>>>(H2, ln3g, ln3b, scal_b, 1024);
  gemm_wmma<0><<<dim3(16, 192), 128, LDSB, stream>>>(V1b_b, flv_b, nullptr, nullptr, Vp, nullptr, 12288, 1024, 512);
  vnorm_kernel<<<8192, 256, 0, stream>>>(Vp, scal_b, 2097152);
  gemm_wmma<1><<<dim3(32, 64), 128, LDSB, stream>>>(scal_b, fw1_b, fb1, nullptr, nullptr, h1_b, 4096, 2048, 1024);
  gemm_wmma<0><<<dim3(16, 64), 128, LDSB, stream>>>(h1_b, fw2_b, fb2, nullptr, so, nullptr, 4096, 1024, 2048);
  final_epilogue<<<8192, 256, 0, stream>>>(so, H2, Vp, V1b, outH, outV, 2097152);
}
